// FDCFormer_30081950941560
// MI455X (gfx1250) — matrix operand order repaired and checked by reading, not compiled, not run on hardware
//
#include <hip/hip_runtime.h>
#include <hip/hip_bf16.h>
#include <math.h>

// ---- problem constants --------------------------------------------------
#define NB     8
#define CDIM   384
#define C3     1152
#define HEADS  8
#define Dh     48      // per-head channels
#define DHh    24
#define D2h    32
#define HW     4096
#define HDIM   64

typedef __bf16 bf16_t;
typedef __attribute__((ext_vector_type(2)))  __bf16 v2bf;
typedef __attribute__((ext_vector_type(16))) __bf16 v16bf;
typedef __attribute__((ext_vector_type(8)))  float  v8f;

// pack two floats -> one 32-bit LDS store of 2 bf16
__device__ __forceinline__ void st_pk(bf16_t* dst, float a, float b) {
  v2bf p; p[0] = (bf16_t)a; p[1] = (bf16_t)b;
  *reinterpret_cast<v2bf*>(dst) = p;
}

// ---- block reduction helpers -------------------------------------------
__device__ __forceinline__ float block_sum(float v, float* red) {
  const int t = threadIdx.x;
  red[t] = v; __syncthreads();
  for (int s = 128; s > 0; s >>= 1) { if (t < s) red[t] += red[t + s]; __syncthreads(); }
  float r = red[0]; __syncthreads();
  return r;
}
__device__ __forceinline__ float block_max(float v, float* red) {
  const int t = threadIdx.x;
  red[t] = v; __syncthreads();
  for (int s = 128; s > 0; s >>= 1) { if (t < s) red[t] = fmaxf(red[t], red[t + s]); __syncthreads(); }
  float r = red[0]; __syncthreads();
  return r;
}

// ---- GEMM tile staging: W[64x64] row-major + X[64k x 128n] transposed -> LDS ----
// Load phase fills registers (clause-able), store phase packs bf16 pairs -> b32 DS stores.
__device__ __forceinline__ void stage64(bf16_t (*sWb)[68], bf16_t (*sXb)[68],
                                        const float* __restrict__ Wq,
                                        const float* __restrict__ Xb,
                                        int oBase, int nBase, int kk, int tid) {
  // W: 64 rows x 64 k ; thread -> (row = tid/4, 16 consecutive k)
  const int o_l  = tid >> 2;            // 0..63
  const int k16  = (tid & 3) * 16;      // 0,16,32,48
  const float* wrow = Wq + (size_t)(oBase + o_l) * CDIM + kk + k16;
  float4 w[4];
#pragma unroll
  for (int j = 0; j < 4; ++j) w[j] = *reinterpret_cast<const float4*>(wrow + 4 * j);
#pragma unroll
  for (int j = 0; j < 4; ++j) {
    st_pk(&sWb[o_l][k16 + 4 * j + 0], w[j].x, w[j].y);
    st_pk(&sWb[o_l][k16 + 4 * j + 2], w[j].z, w[j].w);
  }
  // X: 64 k x 128 n, stored transposed [n][k]; thread loads (k2,n),(k2+1,n) pairs,
  // coalesced across threads in n; writes one b32 per pair.
  float x0[16], x1[16];
#pragma unroll
  for (int j = 0; j < 16; ++j) {
    const int lin = tid + j * 256;      // 0..4095 pairs
    const int n_l = lin & 127;
    const int k2  = (lin >> 7) * 2;     // 0,2,..,62
    const float* xp = Xb + (size_t)(kk + k2) * HW + nBase + n_l;
    x0[j] = xp[0];
    x1[j] = xp[HW];
  }
#pragma unroll
  for (int j = 0; j < 16; ++j) {
    const int lin = tid + j * 256;
    const int n_l = lin & 127;
    const int k2  = (lin >> 7) * 2;
    st_pk(&sXb[n_l][k2], x0[j], x1[j]);
  }
}

// ---- 1) QKV 1x1-conv GEMM: out[b,o,n] = sum_c W[o,c]*X[b,c,n], bf16 WMMA ----
// 256 threads = 8 waves (2x4); block tile 64(o) x 128(n); wave tile 32x32 = 4 WMMA tiles;
// K tile 64 (2 sub-steps -> 8 WMMAs per barrier); double-buffered LDS.
__global__ __launch_bounds__(256) void k_qkv_gemm(const float* __restrict__ X,
                                                  const float* __restrict__ Wq,
                                                  bf16_t* __restrict__ out) {
  __shared__ bf16_t sW[2][64][68];      // [buf][o][k]
  __shared__ bf16_t sX[2][128][68];     // [buf][n][k]

  const int b     = blockIdx.z;
  const int oBase = blockIdx.y * 64;
  const int nBase = blockIdx.x * 128;
  const int tid   = threadIdx.x;
  const int lane  = tid & 31;
  const int wave  = tid >> 5;
  const int wo    = wave >> 2;          // 0..1  -> 32 rows
  const int wn    = wave & 3;           // 0..3  -> 32 cols
  const float* Xb = X + (size_t)b * CDIM * HW;

  // WMMA per-lane layout indices (ISA 7.12.2, wave32)
  const int m    = lane & 15;
  const int kbA  = (lane < 16) ? 0 : 8;    // A: 16-bit 16x32
  const int kbB  = (lane < 16) ? 0 : 8;    // B: 16-bit 32x16
  const int ncol = lane & 15;

  v8f acc00 = {}, acc01 = {}, acc10 = {}, acc11 = {};

  stage64(sW[0], sX[0], Wq, Xb, oBase, nBase, 0, tid);
  __syncthreads();

  int buf = 0;
  for (int kk = 0; kk < CDIM; kk += 64, buf ^= 1) {
    if (kk + 64 < CDIM)
      stage64(sW[buf ^ 1], sX[buf ^ 1], Wq, Xb, oBase, nBase, kk + 64, tid);

#pragma unroll
    for (int s = 0; s < 2; ++s) {
      const int ks = s * 32;
      v16bf a0, a1, b0, b1;
#pragma unroll
      for (int i = 0; i < 8; ++i) {
        a0[i]     = sW[buf][wo * 32 + m][ks + kbA + i];
        a0[8 + i] = sW[buf][wo * 32 + m][ks + 16 + kbA + i];
        a1[i]     = sW[buf][wo * 32 + 16 + m][ks + kbA + i];
        a1[8 + i] = sW[buf][wo * 32 + 16 + m][ks + 16 + kbA + i];
      }
#pragma unroll
      for (int i = 0; i < 16; ++i) {
        b0[i] = sX[buf][wn * 32 + ncol][ks + kbB + (i < 8 ? i : i + 8)];
        b1[i] = sX[buf][wn * 32 + 16 + ncol][ks + kbB + (i < 8 ? i : i + 8)];
      }
      acc00 = __builtin_amdgcn_wmma_f32_16x16x32_bf16(false, a0, false, b0, (short)0, acc00, false, false);
      acc01 = __builtin_amdgcn_wmma_f32_16x16x32_bf16(false, a0, false, b1, (short)0, acc01, false, false);
      acc10 = __builtin_amdgcn_wmma_f32_16x16x32_bf16(false, a1, false, b0, (short)0, acc10, false, false);
      acc11 = __builtin_amdgcn_wmma_f32_16x16x32_bf16(false, a1, false, b1, (short)0, acc11, false, false);
    }
    __syncthreads();
  }

  // D layout: VGPR r -> M = (lane<16 ? r : 8+r), N = lane%16
  const int mb = (lane < 16) ? 0 : 8;
#pragma unroll
  for (int r = 0; r < 8; ++r) {
    const int o0 = oBase + wo * 32 + mb + r;
    const int o1 = o0 + 16;
    const int n0 = nBase + wn * 32 + ncol;
    const int n1 = n0 + 16;
    out[((size_t)b * C3 + o0) * HW + n0] = (bf16_t)acc00[r];
    out[((size_t)b * C3 + o0) * HW + n1] = (bf16_t)acc01[r];
    out[((size_t)b * C3 + o1) * HW + n0] = (bf16_t)acc10[r];
    out[((size_t)b * C3 + o1) * HW + n1] = (bf16_t)acc11[r];
  }
}

// ---- 2) depthwise 3x3, SAME padding: bf16 in, f32 out -------------------
__global__ __launch_bounds__(256) void k_dwconv(const bf16_t* __restrict__ in,
                                                const float* __restrict__ wdw,
                                                float* __restrict__ out) {
  const int b  = blockIdx.z;
  const int ch = blockIdx.y;
  const int p  = blockIdx.x * 256 + threadIdx.x;
  const int y  = p >> 6, x = p & 63;
  const float* wp = wdw + (size_t)ch * 9;
  const bf16_t* ib = in + ((size_t)b * C3 + ch) * HW;
  float acc = 0.f;
#pragma unroll
  for (int dy = -1; dy <= 1; ++dy) {
    const int yy = y + dy;
    if (yy < 0 || yy > 63) continue;
#pragma unroll
    for (int dx = -1; dx <= 1; ++dx) {
      const int xx = x + dx;
      if (xx < 0 || xx > 63) continue;
      acc += wp[(dy + 1) * 3 + (dx + 1)] * (float)ib[yy * 64 + xx];
    }
  }
  out[((size_t)b * C3 + ch) * HW + p] = acc;
}

// ---- 3) attn row: l2norm(q)*l2norm(k)*temp, softmax over hw, in-place over q ----
__global__ __launch_bounds__(256) void k_attn(float* __restrict__ qkv,
                                              const float* __restrict__ temperature) {
  __shared__ float red[256];
  const int r   = blockIdx.x;             // 0..3071 = (b, h*48+d)
  const int b   = r / CDIM;
  const int rem = r % CDIM;               // h*48 + d
  const int h   = rem / Dh;
  float* qrow        = qkv + ((size_t)b * C3 + rem) * HW;
  const float* krow  = qrow + (size_t)CDIM * HW;
  const int tid = threadIdx.x;

  float4 qv[4], kv[4];
  float sq = 0.f, sk = 0.f;
#pragma unroll
  for (int j = 0; j < 4; ++j) {
    const int n = j * 1024 + tid * 4;
    qv[j] = *reinterpret_cast<const float4*>(qrow + n);
    kv[j] = *reinterpret_cast<const float4*>(krow + n);
    sq += qv[j].x * qv[j].x + qv[j].y * qv[j].y + qv[j].z * qv[j].z + qv[j].w * qv[j].w;
    sk += kv[j].x * kv[j].x + kv[j].y * kv[j].y + kv[j].z * kv[j].z + kv[j].w * kv[j].w;
  }
  const float SQ = block_sum(sq, red);
  const float SK = block_sum(sk, red);
  const float scale = (1.f / fmaxf(sqrtf(SQ), 1e-12f)) *
                      (1.f / fmaxf(sqrtf(SK), 1e-12f)) * temperature[h];
  float s[16];
  float mx = -1e30f;
#pragma unroll
  for (int j = 0; j < 4; ++j) {
    s[j * 4 + 0] = qv[j].x * kv[j].x * scale;
    s[j * 4 + 1] = qv[j].y * kv[j].y * scale;
    s[j * 4 + 2] = qv[j].z * kv[j].z * scale;
    s[j * 4 + 3] = qv[j].w * kv[j].w * scale;
#pragma unroll
    for (int i = 0; i < 4; ++i) mx = fmaxf(mx, s[j * 4 + i]);
  }
  const float M = block_max(mx, red);
  float lsum = 0.f;
#pragma unroll
  for (int i = 0; i < 16; ++i) { s[i] = expf(s[i] - M); lsum += s[i]; }
  const float S = block_sum(lsum, red);
  const float inv = 1.f / S;
#pragma unroll
  for (int j = 0; j < 4; ++j) {
    const int n = j * 1024 + tid * 4;
    float4 o;
    o.x = s[j * 4 + 0] * inv; o.y = s[j * 4 + 1] * inv;
    o.z = s[j * 4 + 2] * inv; o.w = s[j * 4 + 3] * inv;
    *reinterpret_cast<float4*>(qrow + n) = o;
  }
}

// ---- 4) fused spatial+channel pool per (b,h) ----------------------------
__global__ __launch_bounds__(256) void k_pool(const float* __restrict__ qkv,
                                              const float* __restrict__ w_ar,
                                              const float* __restrict__ w_vr,
                                              const float* __restrict__ w_al,
                                              const float* __restrict__ w_vl,
                                              float* __restrict__ msp,
                                              float* __restrict__ ctx) {
  __shared__ float red[256];
  __shared__ float abar_s[48];
  __shared__ float vbar_s[48];
  __shared__ float gb_s[24];
  __shared__ float avg_s[24];
  __shared__ float wv_s[48];
  const int bh = blockIdx.x;
  const int b = bh >> 3, h = bh & 7;
  const int tid = threadIdx.x;
  const float* attn = qkv + ((size_t)b * C3 + h * Dh) * HW;             // q region
  const float* v    = qkv + ((size_t)b * C3 + 2 * CDIM + h * Dh) * HW;  // v region

  if (tid < 48) { abar_s[tid] = 0.f; vbar_s[tid] = 0.f; }
  __syncthreads();

  float abar[48], sar[16];
#pragma unroll
  for (int d = 0; d < 48; ++d) abar[d] = 0.f;
#pragma unroll
  for (int j = 0; j < 16; ++j) {
    const int n = j * 256 + tid;
    float s = 0.f;
#pragma unroll
    for (int d = 0; d < 48; ++d) {
      const float av = attn[(size_t)d * HW + n];
      s += w_ar[d] * av;
      abar[d] += av;
    }
    sar[j] = s;
  }
  // cross-thread sum of abar: wave shuffle then LDS atomics
#pragma unroll
  for (int d = 0; d < 48; ++d) {
    float vs = abar[d];
    for (int off = 16; off > 0; off >>= 1) vs += __shfl_xor(vs, off, 32);
    if ((tid & 31) == 0) atomicAdd(&abar_s[d], vs);
  }
  // softmax(cmask) over hw
  float mx = -1e30f;
#pragma unroll
  for (int j = 0; j < 16; ++j) mx = fmaxf(mx, sar[j]);
  const float M = block_max(mx, red);
  float e[16];
  float lsum = 0.f;
#pragma unroll
  for (int j = 0; j < 16; ++j) { e[j] = expf(sar[j] - M); lsum += e[j]; }
  const float S = block_sum(lsum, red);     // barriers here also publish abar_s
  const float cscale = 1.f / S;

  // avg_x = softmax over DH of w_al * mean_n(attn)
  if (tid < 24) {
    float g = 0.f;
#pragma unroll
    for (int d = 0; d < 48; ++d) g += w_al[tid * 48 + d] * abar_s[d];
    gb_s[tid] = g * (1.f / (float)HW);
  }
  __syncthreads();
  if (tid == 0) {
    float mm = -1e30f;
    for (int j = 0; j < 24; ++j) mm = fmaxf(mm, gb_s[j]);
    float ss = 0.f;
    for (int j = 0; j < 24; ++j) { const float ee = expf(gb_s[j] - mm); avg_s[j] = ee; ss += ee; }
    const float iv = 1.f / ss;
    for (int j = 0; j < 24; ++j) avg_s[j] *= iv;
  }
  __syncthreads();
  if (tid < 48) {
    float w = 0.f;
#pragma unroll
    for (int j = 0; j < 24; ++j) w += avg_s[j] * w_vl[j * 48 + tid];
    wv_s[tid] = w;
  }
  __syncthreads();

  // one pass over v: spatial mask + cmask-weighted v average
  float vbar[48];
#pragma unroll
  for (int d = 0; d < 48; ++d) vbar[d] = 0.f;
  float* mrow = msp + (size_t)bh * HW;
#pragma unroll
  for (int j = 0; j < 16; ++j) {
    const int n = j * 256 + tid;
    const float cn = e[j] * cscale;
    float macc = 0.f;
#pragma unroll
    for (int d = 0; d < 48; ++d) {
      const float vv = v[(size_t)d * HW + n];
      macc += wv_s[d] * vv;
      vbar[d] += cn * vv;
    }
    mrow[n] = 1.f / (1.f + expf(-macc));
  }
#pragma unroll
  for (int d = 0; d < 48; ++d) {
    float vs = vbar[d];
    for (int off = 16; off > 0; off >>= 1) vs += __shfl_xor(vs, off, 32);
    if ((tid & 31) == 0) atomicAdd(&vbar_s[d], vs);
  }
  __syncthreads();
  if (tid < 24) {
    float c = 0.f;
#pragma unroll
    for (int d = 0; d < 48; ++d) c += w_vr[tid * 48 + d] * vbar_s[d];
    ctx[(size_t)bh * 24 + tid] = c;
  }
}

// ---- 5) tiny MLP + LayerNorm -> channel gate ----------------------------
__global__ __launch_bounds__(256) void k_mlp(const float* __restrict__ ctx,
                                             const float* __restrict__ w_up1,
                                             const float* __restrict__ b_up1,
                                             const float* __restrict__ ln_g,
                                             const float* __restrict__ ln_b,
                                             const float* __restrict__ w_up2,
                                             const float* __restrict__ b_up2,
                                             float* __restrict__ mch) {
  __shared__ float red[256];
  __shared__ float ctx_s[8][24];
  __shared__ float t_s[32][8];
  const int b = blockIdx.x, tid = threadIdx.x;
  if (tid < 192) ctx_s[tid / 24][tid % 24] = ctx[(size_t)b * 192 + tid];
  __syncthreads();
  const int o2 = tid >> 3, h = tid & 7;
  float t = b_up1[o2];
#pragma unroll
  for (int j = 0; j < 24; ++j) t += w_up1[o2 * 24 + j] * ctx_s[h][j];
  const float S  = block_sum(t, red);
  const float S2 = block_sum(t * t, red);
  const float mu  = S * (1.f / 256.f);
  const float var = S2 * (1.f / 256.f) - mu * mu;
  float tn = (t - mu) * rsqrtf(var + 1e-5f) * ln_g[o2 * 8 + h] + ln_b[o2 * 8 + h];
  tn = fmaxf(tn, 0.f);
  t_s[o2][h] = tn;
  __syncthreads();
  for (int idx = tid; idx < 384; idx += 256) {
    const int d = idx >> 3, hh = idx & 7;
    float u = b_up2[d];
#pragma unroll
    for (int o = 0; o < 32; ++o) u += w_up2[d * 32 + o] * t_s[o][hh];
    mch[(size_t)b * CDIM + d * 8 + hh] = 1.f / (1.f + expf(-u));  // c = d*8 + h
  }
}

// ---- 6) final combine ---------------------------------------------------
__global__ __launch_bounds__(256) void k_out(const float* __restrict__ x,
                                             const float* __restrict__ mch,
                                             const float* __restrict__ msp,
                                             const float* __restrict__ w_proj,
                                             float* __restrict__ out) {
  const int b = blockIdx.z, c = blockIdx.y;
  const int n = blockIdx.x * 256 + threadIdx.x;
  const size_t base = ((size_t)b * CDIM + c) * HW + n;
  float wp[8];
#pragma unroll
  for (int h = 0; h < 8; ++h) wp[h] = w_proj[c * 8 + h];
  const float g = mch[b * CDIM + c];
  float acc = 0.f;
#pragma unroll
  for (int h = 0; h < 8; ++h) acc += wp[h] * msp[((size_t)b * 8 + h) * HW + n];
  out[base] = x[base] * (g + acc);
}

// ---- launcher -----------------------------------------------------------
extern "C" void kernel_launch(void* const* d_in, const int* in_sizes, int n_in,
                              void* d_out, int out_size, void* d_ws, size_t ws_size,
                              hipStream_t stream) {
  (void)in_sizes; (void)n_in; (void)out_size; (void)ws_size;
  const float* x      = (const float*)d_in[0];
  const float* w_qkv  = (const float*)d_in[1];
  const float* w_dw   = (const float*)d_in[2];
  const float* temp   = (const float*)d_in[3];
  const float* w_ar   = (const float*)d_in[4];
  const float* w_vr   = (const float*)d_in[5];
  const float* w_up1  = (const float*)d_in[6];
  const float* b_up1  = (const float*)d_in[7];
  const float* ln_g   = (const float*)d_in[8];
  const float* ln_b   = (const float*)d_in[9];
  const float* w_up2  = (const float*)d_in[10];
  const float* b_up2  = (const float*)d_in[11];
  const float* w_al   = (const float*)d_in[12];
  const float* w_vl   = (const float*)d_in[13];
  const float* w_proj = (const float*)d_in[14];
  float* out = (float*)d_out;

  const size_t NQKV = (size_t)NB * C3 * HW;   // 37,748,736
  char* p = (char*)d_ws;
  bf16_t* qkv_pre = (bf16_t*)p;  p += NQKV * sizeof(bf16_t);     // 75.5 MB
  float*  qkv     = (float*)p;   p += NQKV * sizeof(float);      // 151 MB (attn overwrites q)
  float*  msp     = (float*)p;   p += (size_t)NB * HEADS * HW * sizeof(float);
  float*  ctx     = (float*)p;   p += (size_t)NB * HEADS * DHh * sizeof(float);
  float*  mch     = (float*)p;   p += (size_t)NB * CDIM * sizeof(float);

  k_qkv_gemm<<<dim3(HW / 128, C3 / 64, NB), 256, 0, stream>>>(x, w_qkv, qkv_pre);
  k_dwconv <<<dim3(HW / 256, C3, NB),       256, 0, stream>>>(qkv_pre, w_dw, qkv);
  k_attn   <<<dim3(NB * CDIM),              256, 0, stream>>>(qkv, temp);
  k_pool   <<<dim3(NB * HEADS),             256, 0, stream>>>(qkv, w_ar, w_vr, w_al, w_vl, msp, ctx);
  k_mlp    <<<dim3(NB),                     256, 0, stream>>>(ctx, w_up1, b_up1, ln_g, ln_b, w_up2, b_up2, mch);
  k_out    <<<dim3(HW / 256, CDIM, NB),     256, 0, stream>>>(x, mch, msp, w_proj, out);
}
